// GATModel_47691316855138
// MI455X (gfx1250) — matrix operand order repaired and checked by reading, not compiled, not run on hardware
//
#include <hip/hip_runtime.h>
#include <hip/hip_bf16.h>

typedef __attribute__((ext_vector_type(16))) _Float16 v16h;
typedef __attribute__((ext_vector_type(8)))  _Float16 v8h;
typedef __attribute__((ext_vector_type(8)))  float    v8f;

#define DEV static __device__ __forceinline__

// order-preserving float -> uint transform for atomic max over signed floats
DEV unsigned ordf(float f) {
  unsigned u = __float_as_uint(f);
  return (u & 0x80000000u) ? ~u : (u | 0x80000000u);
}
DEV float deordf(unsigned u) {
  return (u & 0x80000000u) ? __uint_as_float(u & 0x7fffffffu)
                           : __uint_as_float(~u);
}
DEV float elu1(float v) { return v > 0.f ? v : (expf(v) - 1.f); }

// ---------------------------------------------------------------- utilities
__global__ void k_zero(unsigned* __restrict__ p, long n) {
  long i  = blockIdx.x * (long)blockDim.x + threadIdx.x;
  long st = gridDim.x * (long)blockDim.x;
  for (; i < n; i += st) p[i] = 0u;
}

// dst[nc*K + k] = (f16) src[k*NC + nc]   (transpose + downconvert weights)
__global__ void k_transpose_f16(const float* __restrict__ src, _Float16* __restrict__ dst,
                                int K, int NC) {
  int i = blockIdx.x * blockDim.x + threadIdx.x;
  if (i >= K * NC) return;
  int nc = i / K, k = i - nc * K;
  dst[i] = (_Float16)src[k * NC + nc];
}

// BatchNorm (eval) + downconvert to f16, row length 128
__global__ void k_bn(const float* __restrict__ x, const float* __restrict__ g,
                     const float* __restrict__ b, const float* __restrict__ m,
                     const float* __restrict__ v, _Float16* __restrict__ h0, long total) {
  long i = blockIdx.x * (long)blockDim.x + threadIdx.x;
  if (i >= total) return;
  int c = (int)(i & 127);
  float val = (x[i] - m[c]) * rsqrtf(v[c] + 1e-5f) * g[c] + b[c];
  h0[i] = (_Float16)val;
}

// ---------------------------------------------------------------- WMMA GEMM
// C[M x NC] = A[M x K](f16,row-major) * B (given as Bt[NC x K], f16 row-major).
// One wave computes a 16-row x 64-col strip: 4 accumulators, K-loop step 32.
__global__ void k_gemm_wmma(const _Float16* __restrict__ A, const _Float16* __restrict__ Bt,
                            float* __restrict__ C, int M, int K, int NC) {
  int wid  = (blockIdx.x * blockDim.x + threadIdx.x) >> 5;
  int lane = threadIdx.x & 31;
  int cg = NC >> 6;
  int ntiles = ((M + 15) >> 4) * cg;
  if (wid >= ntiles) return;
  int row0 = (wid / cg) << 4;
  int col0 = (wid % cg) << 6;
  int half = lane >> 4, mn = lane & 15;

  int rowa = row0 + mn; if (rowa >= M) rowa = M - 1;
  const _Float16* arow = A + (long)rowa * K + (half << 4);          // K = 16*half + ...
  const _Float16* b0 = Bt + (long)(col0 + mn) * K + (half << 4);    // K = 16*half + 0..15
  const _Float16* b1p = b0 + (long)16 * K;
  const _Float16* b2p = b0 + (long)32 * K;
  const _Float16* b3p = b0 + (long)48 * K;

  v8f acc0 = {}, acc1 = {}, acc2 = {}, acc3 = {};
  for (int kb = 0; kb < K; kb += 32) {
    __builtin_prefetch(arow + kb + 64, 0, 1);
    union { v16h v; v8h h[2]; } a;
    a.h[0] = *(const v8h*)(arow + kb);        // K = kb + 16*half + 0..7
    a.h[1] = *(const v8h*)(arow + kb + 8);    // K = kb + 16*half + 8..15
    v16h bf;
    bf = *(const v16h*)(b0 + kb);
    acc0 = __builtin_amdgcn_wmma_f32_16x16x32_f16(false, a.v, false, bf, (short)0, acc0, false, false);
    bf = *(const v16h*)(b1p + kb);
    acc1 = __builtin_amdgcn_wmma_f32_16x16x32_f16(false, a.v, false, bf, (short)0, acc1, false, false);
    bf = *(const v16h*)(b2p + kb);
    acc2 = __builtin_amdgcn_wmma_f32_16x16x32_f16(false, a.v, false, bf, (short)0, acc2, false, false);
    bf = *(const v16h*)(b3p + kb);
    acc3 = __builtin_amdgcn_wmma_f32_16x16x32_f16(false, a.v, false, bf, (short)0, acc3, false, false);
  }
  // C/D layout: VGPR r -> M = row0 + 8*half + r, N = col0 + lane&15 (+16*t)
  float* crow = C + (long)(row0 + (half << 3)) * NC + col0 + mn;
  for (int r = 0; r < 8; r++) {
    int rr = row0 + (half << 3) + r;
    if (rr < M) {
      crow[0]  = acc0[r];
      crow[16] = acc1[r];
      crow[32] = acc2[r];
      crow[48] = acc3[r];
    }
    crow += NC;
  }
}

// ---------------------------------------------------------------- attention
// per (node, head): a_s = <feat, att_src[h]>, a_d = <feat, att_dst[h]>, ch=64
__global__ void k_scores(const float* __restrict__ feat, const float* __restrict__ asrc,
                         const float* __restrict__ adst, float* __restrict__ outs,
                         float* __restrict__ outd, int N, int H) {
  int i = blockIdx.x * blockDim.x + threadIdx.x;
  if (i >= N * H) return;
  int h = i % H;
  const float* f  = feat + (long)i * 64;
  const float* as = asrc + h * 64;
  const float* ad = adst + h * 64;
  float s = 0.f, d = 0.f;
  for (int c = 0; c < 64; c++) { float v = f[c]; s += v * as[c]; d += v * ad[c]; }
  outs[i] = s; outd[i] = d;
}

// pass A: segment max of leaky_relu(a_s[src] + a_d[dst]) per (dst, head)
__global__ void k_edge_max(const int* __restrict__ ei, const float* __restrict__ as,
                           const float* __restrict__ ad, unsigned* __restrict__ amax,
                           int E, int ET, int H) {
  long idx   = blockIdx.x * (long)blockDim.x + threadIdx.x;
  long total = (long)ET * H;
  long st    = gridDim.x * (long)blockDim.x;
  for (; idx < total; idx += st) {
    int e = (int)(idx / H), h = (int)(idx - (long)e * H);
    int s = (e < E) ? ei[e]     : (e - E);
    int d = (e < E) ? ei[E + e] : (e - E);
    float ev = as[s * H + h] + ad[d * H + h];
    ev = ev > 0.f ? ev : 0.2f * ev;
    atomicMax(&amax[d * H + h], ordf(ev));
  }
}

// pass B: one wave per (edge, head); accumulate exp and exp-weighted features
__global__ void k_edge_agg(const int* __restrict__ ei, const float* __restrict__ as,
                           const float* __restrict__ ad, const unsigned* __restrict__ amax,
                           float* __restrict__ denom, const float* __restrict__ feat,
                           float* __restrict__ num, int E, int ET, int H) {
  int lane = threadIdx.x & 31;
  long wid = (blockIdx.x * (long)blockDim.x + threadIdx.x) >> 5;
  long nw  = (gridDim.x * (long)blockDim.x) >> 5;
  long total = (long)ET * H;
  for (long t = wid; t < total; t += nw) {
    int e = (int)(t / H), h = (int)(t - (long)e * H);
    int s = (e < E) ? ei[e]     : (e - E);
    int d = (e < E) ? ei[E + e] : (e - E);
    float ev = as[s * H + h] + ad[d * H + h];
    ev = ev > 0.f ? ev : 0.2f * ev;
    float ex = expf(ev - deordf(amax[d * H + h]));
    if (lane == 0) atomicAdd(&denom[d * H + h], ex);
    const float* fs = feat + ((long)s * H + h) * 64 + lane * 2;
    float*       nd = num  + ((long)d * H + h) * 64 + lane * 2;
    atomicAdd(nd,     ex * fs[0]);
    atomicAdd(nd + 1, ex * fs[1]);
  }
}

// pass C: out = ELU(num/denom + bias); optionally emit f16 (next GEMM) or f32
__global__ void k_finalize(const float* __restrict__ num, const float* __restrict__ denom,
                           const float* __restrict__ bias, _Float16* __restrict__ oh,
                           float* __restrict__ of, int N, int H) {
  long i = blockIdx.x * (long)blockDim.x + threadIdx.x;
  long total = (long)N * H * 64;
  if (i >= total) return;
  int rem = (int)(i % (H * 64));
  long nh = i >> 6;  // n*H + h
  float v = num[i] / (denom[nh] + 1e-16f) + bias[rem];
  v = elu1(v);
  if (oh) oh[i] = (_Float16)v;
  if (of) of[i] = v;
}

// ---------------------------------------------------------------- head
__global__ void k_head(const float* __restrict__ h2, const int* __restrict__ drug_idx,
                       const int* __restrict__ dis_idx, const int* __restrict__ drug_id,
                       const int* __restrict__ dis_id, const float* __restrict__ drug_tab,
                       const float* __restrict__ dis_tab, const float* __restrict__ cW1,
                       const float* __restrict__ cb1, const float* __restrict__ cW2,
                       const float* __restrict__ cb2, float* __restrict__ out) {
  __shared__ float pooled[256];
  __shared__ float hid[64];
  int t = threadIdx.x;
  if (t < 64) {
    float s = 0.f;
    for (int i = 0; i < 64; i++) s += h2[(long)drug_idx[i] * 64 + t];
    pooled[t] = s * (1.f / 64.f);
  } else if (t < 128) {
    int c = t - 64; float s = 0.f;
    for (int i = 0; i < 64; i++) s += h2[(long)dis_idx[i] * 64 + c];
    pooled[t] = s * (1.f / 64.f);
  } else if (t < 192) {
    pooled[t] = drug_tab[(long)drug_id[0] * 64 + (t - 128)];
  } else {
    pooled[t] = dis_tab[(long)dis_id[0] * 64 + (t - 192)];
  }
  __syncthreads();
  if (t < 64) {
    float s = cb1[t];
    for (int i = 0; i < 256; i++) s += pooled[i] * cW1[i * 64 + t];
    hid[t] = elu1(s);
  }
  __syncthreads();
  if (t == 0) {
    float s = cb2[0];
    for (int j = 0; j < 64; j++) s += hid[j] * cW2[j];
    out[0] = s;
  }
}

// ---------------------------------------------------------------- launch
extern "C" void kernel_launch(void* const* d_in, const int* in_sizes, int n_in,
                              void* d_out, int out_size, void* d_ws, size_t ws_size,
                              hipStream_t stream) {
  const float* x        = (const float*)d_in[0];
  const int*   ei       = (const int*)d_in[1];
  const int*   drug_gi  = (const int*)d_in[2];
  const int*   dis_gi   = (const int*)d_in[3];
  const int*   drug_id  = (const int*)d_in[4];
  const int*   dis_id   = (const int*)d_in[5];
  const float* bn_g     = (const float*)d_in[6];
  const float* bn_b     = (const float*)d_in[7];
  const float* bn_m     = (const float*)d_in[8];
  const float* bn_v     = (const float*)d_in[9];
  const float* W1       = (const float*)d_in[10];
  const float* att_s1   = (const float*)d_in[11];
  const float* att_d1   = (const float*)d_in[12];
  const float* b1       = (const float*)d_in[13];
  const float* W2       = (const float*)d_in[14];
  const float* att_s2   = (const float*)d_in[15];
  const float* att_d2   = (const float*)d_in[16];
  const float* b2       = (const float*)d_in[17];
  const float* drug_tab = (const float*)d_in[18];
  const float* dis_tab  = (const float*)d_in[19];
  const float* cW1      = (const float*)d_in[20];
  const float* cb1      = (const float*)d_in[21];
  const float* cW2      = (const float*)d_in[22];
  const float* cb2      = (const float*)d_in[23];

  const int N  = in_sizes[0] / 128;     // 50000
  const int E  = in_sizes[1] / 2;       // 800000
  const int ET = E + N;                 // + self loops
  const int H1 = 4;

  // workspace carve-out (256B aligned)
  char* p = (char*)d_ws;
  auto carve = [&](size_t bytes) -> void* {
    void* r = (void*)p;
    p += (bytes + 255) & ~(size_t)255;
    return r;
  };
  _Float16* wt1    = (_Float16*)carve((size_t)256 * 128 * 2);
  _Float16* wt2    = (_Float16*)carve((size_t)64 * 256 * 2);
  _Float16* h0     = (_Float16*)carve((size_t)N * 128 * 2);
  float*    feat1  = (float*)carve((size_t)N * 256 * 4);
  float*    as1    = (float*)carve((size_t)N * H1 * 4);
  float*    ad1    = (float*)carve((size_t)N * H1 * 4);
  unsigned* amax1  = (unsigned*)carve((size_t)N * H1 * 4);
  float*    den1   = (float*)carve((size_t)N * H1 * 4);
  float*    num1   = (float*)carve((size_t)N * 256 * 4);
  _Float16* h1f16  = (_Float16*)carve((size_t)N * 256 * 2);
  float*    feat2  = (float*)carve((size_t)N * 64 * 4);
  float*    as2    = (float*)carve((size_t)N * 4);
  float*    ad2    = (float*)carve((size_t)N * 4);
  unsigned* amax2  = (unsigned*)carve((size_t)N * 4);
  float*    den2   = (float*)carve((size_t)N * 4);
  float*    num2   = (float*)carve((size_t)N * 64 * 4);
  float*    h2     = (float*)carve((size_t)N * 64 * 4);

  const int BT = 256;
  auto cdiv = [](long a, long b) { return (int)((a + b - 1) / b); };

  // weight transpose + downconvert
  k_transpose_f16<<<cdiv(128 * 256, BT), BT, 0, stream>>>(W1, wt1, 128, 256);
  k_transpose_f16<<<cdiv(256 * 64, BT), BT, 0, stream>>>(W2, wt2, 256, 64);

  // BatchNorm -> f16
  k_bn<<<cdiv((long)N * 128, BT), BT, 0, stream>>>(x, bn_g, bn_b, bn_m, bn_v, h0, (long)N * 128);

  // ---- layer 1 ----
  {
    int ntiles = ((N + 15) / 16) * (256 / 64);
    k_gemm_wmma<<<cdiv((long)ntiles * 32, BT), BT, 0, stream>>>(h0, wt1, feat1, N, 128, 256);
  }
  k_scores<<<cdiv((long)N * H1, BT), BT, 0, stream>>>(feat1, att_s1, att_d1, as1, ad1, N, H1);
  k_zero<<<1024, BT, 0, stream>>>(amax1, (long)N * H1);
  k_zero<<<1024, BT, 0, stream>>>((unsigned*)den1, (long)N * H1);
  k_zero<<<4096, BT, 0, stream>>>((unsigned*)num1, (long)N * 256);
  k_edge_max<<<8192, BT, 0, stream>>>(ei, as1, ad1, amax1, E, ET, H1);
  k_edge_agg<<<4096, BT, 0, stream>>>(ei, as1, ad1, amax1, den1, feat1, num1, E, ET, H1);
  k_finalize<<<cdiv((long)N * 256, BT), BT, 0, stream>>>(num1, den1, b1, h1f16, nullptr, N, H1);

  // ---- layer 2 ----
  {
    int ntiles = ((N + 15) / 16) * (64 / 64);
    k_gemm_wmma<<<cdiv((long)ntiles * 32, BT), BT, 0, stream>>>(h1f16, wt2, feat2, N, 256, 64);
  }
  k_scores<<<cdiv((long)N, BT), BT, 0, stream>>>(feat2, att_s2, att_d2, as2, ad2, N, 1);
  k_zero<<<1024, BT, 0, stream>>>(amax2, (long)N);
  k_zero<<<1024, BT, 0, stream>>>((unsigned*)den2, (long)N);
  k_zero<<<4096, BT, 0, stream>>>((unsigned*)num2, (long)N * 64);
  k_edge_max<<<8192, BT, 0, stream>>>(ei, as2, ad2, amax2, E, ET, 1);
  k_edge_agg<<<4096, BT, 0, stream>>>(ei, as2, ad2, amax2, den2, feat2, num2, E, ET, 1);
  k_finalize<<<cdiv((long)N * 64, BT), BT, 0, stream>>>(num2, den2, b2, nullptr, h2, N, 1);

  // ---- head ----
  k_head<<<1, 256, 0, stream>>>(h2, drug_gi, dis_gi, drug_id, dis_id,
                                drug_tab, dis_tab, cW1, cb1, cW2, cb2, (float*)d_out);
}
